// TopKEdges_25786983645321
// MI455X (gfx1250) — matrix operand order repaired and checked by reading, not compiled, not run on hardware
//
#include <hip/hip_runtime.h>
#include <hip/hip_bf16.h>
#include <math.h>

// ---------------------------------------------------------------------------
// Shapes (fixed by the reference): n=32, ma=512, nfs=64, hid=128, out=16, K=32
// ---------------------------------------------------------------------------
#define NB    32
#define MA    512
#define NFS   64
#define HID   128
#define NOUT  16
#define TOPK  32
#define RCUT  5.0f

typedef __attribute__((ext_vector_type(16))) _Float16 v16h;
typedef __attribute__((ext_vector_type(8)))  _Float16 v8h;
typedef __attribute__((ext_vector_type(8)))  float    v8f;

// ---------------------------------------------------------------------------
// Kernel 0: copy z (as float) and r into d_out; build f16 transposed weights.
//   W1T[n][k] = (f16) W1[k][n]   (128 x 128)
//   W2T[n][k] = (f16) W2[k][n]   (16  x 128)
// ---------------------------------------------------------------------------
__global__ __launch_bounds__(256)
void prep_kernel(const int* __restrict__ z, const float* __restrict__ r,
                 const float* __restrict__ W1, const float* __restrict__ W2,
                 float* __restrict__ outz, float* __restrict__ outr,
                 _Float16* __restrict__ W1T, _Float16* __restrict__ W2T) {
  int idx = blockIdx.x * blockDim.x + threadIdx.x;           // 0 .. 49151
  if (idx < NB * MA) outz[idx] = (float)z[idx];
  if (idx < NB * MA * 3) outr[idx] = r[idx];
  if (idx < HID * HID) {                                     // 16384
    int n = idx >> 7, k = idx & 127;
    W1T[n * HID + k] = (_Float16)W1[k * HID + n];
  }
  if (idx < NOUT * HID) {                                    // 2048
    int n = idx >> 7, k = idx & 127;
    W2T[n * HID + k] = (_Float16)W2[k * NOUT + n];
  }
}

// ---------------------------------------------------------------------------
// Kernel 1: cutoff matrix + top-K.  One wave32 per (b,i) row of 512 candidates.
// Each lane holds 16 candidates in registers; 32 rounds of wave-wide argmax
// (ties -> smaller index, matching jax.lax.top_k ordering).
// ---------------------------------------------------------------------------
__global__ __launch_bounds__(256)
void topk_kernel(const int* __restrict__ z, const float* __restrict__ r,
                 float* __restrict__ vals, int* __restrict__ idxs) {
  const int row  = (blockIdx.x * blockDim.x + threadIdx.x) >> 5;  // b*MA + i
  const int lane = threadIdx.x & 31;
  const int b = row >> 9, i = row & (MA - 1);

  const float* rb = r + (size_t)b * MA * 3;
  const int*   zb = z + (size_t)b * MA;
  const float rix = rb[i * 3 + 0], riy = rb[i * 3 + 1], riz = rb[i * 3 + 2];
  const float mi  = (zb[i] > -1) ? 1.0f : 0.0f;
  const float pi_over_rc = 3.14159265358979f / RCUT;

  float cand[16];
#pragma unroll
  for (int t = 0; t < 16; ++t) {
    int j = lane + 32 * t;
    float dx = rb[j * 3 + 0] - rix;
    float dy = rb[j * 3 + 1] - riy;
    float dz = rb[j * 3 + 2] - riz;
    float d  = sqrtf(dx * dx + dy * dy + dz * dz);
    float cf = 0.5f * (__cosf(fminf(d, RCUT) * pi_over_rc) + 1.0f);
    float mj = (zb[j] > -1) ? 1.0f : 0.0f;
    cf *= mi * mj;
    if (j == i) cf = 0.0f;        // (cf - eye) * (mask - eye) => diagonal = 0
    cand[t] = cf;
  }

  float myv = 0.0f; int myj = 0;
  for (int rr = 0; rr < TOPK; ++rr) {
    // local argmax (strict > keeps smallest index on ties)
    float bv = cand[0]; int bt = 0;
#pragma unroll
    for (int t = 1; t < 16; ++t)
      if (cand[t] > bv) { bv = cand[t]; bt = t; }
    int bj = lane + 32 * bt;
    // wave32 butterfly argmax with index tie-break
#pragma unroll
    for (int off = 16; off > 0; off >>= 1) {
      float ov = __shfl_xor(bv, off, 32);
      int   oj = __shfl_xor(bj, off, 32);
      if (ov > bv || (ov == bv && oj < bj)) { bv = ov; bj = oj; }
    }
    if (lane == rr) { myv = bv; myj = bj; }
    if (lane == (bj & 31)) cand[bj >> 5] = -1e30f;   // retire winner
  }
  vals[(size_t)row * TOPK + lane] = myv;
  idxs[(size_t)row * TOPK + lane] = myj;
}

// ---------------------------------------------------------------------------
// Kernel 2: fused edge MLP via WMMA.  One wave32 per 16-edge tile
// (tile = (b, i, khalf); rows m=0..15 <-> neighbor slots khalf*16+m).
//   GEMM1: u[16x128] @ W1[128x128] -> silu -> act[16x128]  (8 Ntiles x 4 Ksteps)
//   GEMM2: act[16x128] @ W2[128x16] -> c[16x16]            (4 Ksteps)
//   out:   c[m][n] * dvn[m][0..2]  -> contiguous 192B per edge row
// Fragments use contiguous k order (k = 16*h + s) for both A and B.
// ---------------------------------------------------------------------------
__global__ __launch_bounds__(128)
void mlp_wmma_kernel(const float* __restrict__ h, const float* __restrict__ r,
                     const float* __restrict__ vals, const int* __restrict__ idxs,
                     const _Float16* __restrict__ W1T, const _Float16* __restrict__ W2T,
                     const float* __restrict__ b1, const float* __restrict__ b2,
                     float* __restrict__ outc) {
  __shared__ __attribute__((aligned(32))) _Float16 lds_u[4][16 * HID];
  __shared__ __attribute__((aligned(32))) _Float16 lds_act[4][16 * HID];
  __shared__ float lds_dvn[4][16 * 3];
  __shared__ float lds_cf[4][16];
  __shared__ int   lds_j[4][16];

  const int w    = threadIdx.x >> 5;
  const int lane = threadIdx.x & 31;
  const int tile = blockIdx.x * 4 + w;
  const int khalf = tile & 1;
  const int row   = tile >> 1;              // b*MA + i
  const int b = row >> 9, i = row & (MA - 1);

  // ---- phase 0a: per-edge neighbor id, cutoff, normalized direction --------
  if (lane < 16) {
    int kk = khalf * 16 + lane;
    int j  = idxs[(size_t)row * TOPK + kk];
    float cfv = vals[(size_t)row * TOPK + kk];
    lds_j[w][lane]  = j;
    lds_cf[w][lane] = cfv;
    const float* rb = r + (size_t)b * MA * 3;
    float dx = rb[j * 3 + 0] - rb[i * 3 + 0];
    float dy = rb[j * 3 + 1] - rb[i * 3 + 1];
    float dz = rb[j * 3 + 2] - rb[i * 3 + 2];
    float nrm = fmaxf(sqrtf(dx * dx + dy * dy + dz * dz), 1e-4f);
    lds_dvn[w][lane * 3 + 0] = dx / nrm;
    lds_dvn[w][lane * 3 + 1] = dy / nrm;
    lds_dvn[w][lane * 3 + 2] = dz / nrm;
  }
  __syncthreads();

  // ---- phase 0b: build u tile [16 x 128] f16: [h_i | h_j * cf] -------------
  const float* hi = h + (size_t)row * NFS;
  for (int m = 0; m < 16; ++m) {
    int jm = lds_j[w][m];
    float cfv = lds_cf[w][m];
    const float* hj = h + (size_t)(b * MA + jm) * NFS;
#pragma unroll
    for (int s = 0; s < 4; ++s) {
      int c = lane + 32 * s;
      float v = (c < NFS) ? hi[c] : hj[c - NFS] * cfv;
      lds_u[w][m * HID + c] = (_Float16)v;
    }
  }
  __syncthreads();

  // ---- fragment addressing (contiguous k = 16*h + s, wave32) ---------------------------
  const int mrow = lane & 15;                // A-matrix row owned by this lane
  const int alo  = (lane < 16) ? 0 : 16;     // A: lo lanes K=0..15, hi K=16..31
  const int ncol = lane & 15;                // B/D column
  const int bko  = (lane < 16) ? 0 : 16;     // B: lo lanes K=0..15, hi K=16..31
  const int mbase = (lane >> 4) << 3;        // D: lo lanes M=v, hi lanes M=v+8

  // A fragments for GEMM1 (4 K-steps of 16x32)
  v16h A1[4];
#pragma unroll
  for (int ks = 0; ks < 4; ++ks) {
    union { v16h v; v8h p[2]; } au;
    au.p[0] = *(const v8h*)&lds_u[w][mrow * HID + ks * 32 + alo];
    au.p[1] = *(const v8h*)&lds_u[w][mrow * HID + ks * 32 + alo + 8];
    A1[ks] = au.v;
  }

  // ---- GEMM1: 8 N-tiles x 4 K-steps, b1-init, SiLU, spill act to LDS -------
#pragma unroll
  for (int nt = 0; nt < 8; ++nt) {
    float b1v = b1[nt * 16 + ncol];
    v8f acc;
#pragma unroll
    for (int v = 0; v < 8; ++v) acc[v] = b1v;
#pragma unroll
    for (int ks = 0; ks < 4; ++ks) {
      v16h B = *(const v16h*)(W1T + (size_t)(nt * 16 + ncol) * HID + ks * 32 + bko);
      acc = __builtin_amdgcn_wmma_f32_16x16x32_f16(false, A1[ks], false, B,
                                                   (short)0, acc, false, false);
    }
#pragma unroll
    for (int v = 0; v < 8; ++v) {
      float x = acc[v];
      float s = x / (1.0f + __expf(-x));     // SiLU
      lds_act[w][(mbase + v) * HID + nt * 16 + ncol] = (_Float16)s;
    }
  }
  __syncthreads();

  // ---- GEMM2: act[16x128] @ W2[128x16] -------------------------------------
  float b2v = b2[ncol];
  v8f acc2;
#pragma unroll
  for (int v = 0; v < 8; ++v) acc2[v] = b2v;
#pragma unroll
  for (int ks = 0; ks < 4; ++ks) {
    union { v16h v; v8h p[2]; } au;
    au.p[0] = *(const v8h*)&lds_act[w][mrow * HID + ks * 32 + alo];
    au.p[1] = *(const v8h*)&lds_act[w][mrow * HID + ks * 32 + alo + 8];
    v16h B = *(const v16h*)(W2T + (size_t)ncol * HID + ks * 32 + bko);
    acc2 = __builtin_amdgcn_wmma_f32_16x16x32_f16(false, au.v, false, B,
                                                  (short)0, acc2, false, false);
  }

  // ---- scale by direction and store c[e][n][0..2] (contiguous per edge) ----
#pragma unroll
  for (int v = 0; v < 8; ++v) {
    int m = mbase + v;
    size_t e = (size_t)row * TOPK + khalf * 16 + m;
    float val = acc2[v];
    float* o = outc + e * (NOUT * 3) + ncol * 3;
    o[0] = val * lds_dvn[w][m * 3 + 0];
    o[1] = val * lds_dvn[w][m * 3 + 1];
    o[2] = val * lds_dvn[w][m * 3 + 2];
  }
}

// ---------------------------------------------------------------------------
// Launch.  Inputs: z, r, h, W1, b1, W2, b2.
// d_out = [z:16384][r:49152][c:25165824] floats.
// Workspace: vals(2MB) | idxs(2MB) | W1T f16(32KB) | W2T f16(4KB).
// ---------------------------------------------------------------------------
extern "C" void kernel_launch(void* const* d_in, const int* in_sizes, int n_in,
                              void* d_out, int out_size, void* d_ws, size_t ws_size,
                              hipStream_t stream) {
  const int*   z  = (const int*)  d_in[0];
  const float* r  = (const float*)d_in[1];
  const float* h  = (const float*)d_in[2];
  const float* W1 = (const float*)d_in[3];
  const float* b1 = (const float*)d_in[4];
  const float* W2 = (const float*)d_in[5];
  const float* b2 = (const float*)d_in[6];

  char* ws = (char*)d_ws;
  float*    vals = (float*)   (ws + 0);
  int*      idxs = (int*)     (ws + (size_t)2 * 1024 * 1024);
  _Float16* W1T  = (_Float16*)(ws + (size_t)4 * 1024 * 1024);
  _Float16* W2T  = (_Float16*)(ws + (size_t)4 * 1024 * 1024 + 32 * 1024);

  float* outz = (float*)d_out;
  float* outr = outz + NB * MA;
  float* outc = outr + NB * MA * 3;

  // 0) weight prep + passthrough outputs (49152 threads covers all sections)
  prep_kernel<<<192, 256, 0, stream>>>(z, r, W1, W2, outz, outr, W1T, W2T);

  // 1) top-K: 16384 rows, one wave per row, 8 waves/block
  topk_kernel<<<(NB * MA) / 8, 256, 0, stream>>>(z, r, vals, idxs);

  // 2) fused WMMA edge-MLP: 32768 tiles, 4 waves/block
  mlp_wmma_kernel<<<(NB * MA * 2) / 4, 128, 0, stream>>>(h, r, vals, idxs,
                                                         W1T, W2T, b1, b2, outc);
}
